// GlobalLocalFilter_46686294507950
// MI455X (gfx1250) — matrix operand order repaired and checked by reading, not compiled, not run on hardware
//
#include <hip/hip_runtime.h>
#include <hip/hip_bf16.h>
#include <math.h>

// ---------------------------------------------------------------------------
// GlobalLocalFilter fused pipeline for gfx1250 (MI455X), wave32 + WMMA + TDM.
//   x:(128,768,14,14) f32 -> dwconv(half1) | circular-conv(half2, from iDFT of
//   complex weight) -> f16 feature matrix [768][25088] -> WMMA f16 GEMM with
//   768x768 projection (fp32 accum, A-tiles staged by Tensor Data Mover)
//   + bias -> fused BN stats -> BN apply.
// ---------------------------------------------------------------------------

typedef __attribute__((ext_vector_type(16))) _Float16     v16h;
typedef __attribute__((ext_vector_type(8)))  _Float16     h8;
typedef __attribute__((ext_vector_type(8)))  float        v8f;
typedef __attribute__((ext_vector_type(4)))  unsigned int u32x4;
typedef __attribute__((ext_vector_type(8)))  int          i32x8;
typedef __attribute__((ext_vector_type(4)))  int          i32x4;

#define B_IMG   128
#define DIM     768
#define HALF    384
#define HW      196            // 14*14
#define P_TOT   25088          // B_IMG * HW
#define K_DIM   768

#if defined(__HIP_DEVICE_COMPILE__) && \
    __has_builtin(__builtin_amdgcn_tensor_load_to_lds) && \
    __has_builtin(__builtin_amdgcn_s_wait_tensorcnt)
#define HAVE_TDM 1
#else
#define HAVE_TDM 0
#endif

// ------------------------------ small kernels ------------------------------

__global__ void init_stats_kernel(float* gsum, float* gsq) {
    int i = blockIdx.x * 256 + threadIdx.x;
    if (i < DIM) { gsum[i] = 0.f; gsq[i] = 0.f; }
}

__global__ void convert_wproj_kernel(const float* __restrict__ w,
                                     _Float16* __restrict__ wh) {
    int i = blockIdx.x * 256 + threadIdx.x;      // DIM*K_DIM exact multiple
    wh[i] = (_Float16)w[i];
}

// k[c][u][v] = (1/196) * sum_{a=0..13,b=0..7} w_b * (Wr*cos(phi) - Wi*sin(phi)),
// phi = 2*pi*(a*u + b*v)/14 ; w_b = 1 for b in {0,7}, 2 otherwise.
// This is exactly irfft2(W, s=(14,14), norm='backward') (c2r convention).
__global__ void build_spatial_kernel(const float* __restrict__ cw,
                                     float* __restrict__ kspat) {
    int idx = blockIdx.x * 256 + threadIdx.x;       // c*196 + pos
    if (idx >= HALF * HW) return;
    int c = idx / HW;
    int pos = idx % HW;
    int u = pos / 14, v = pos % 14;
    const float* wc = cw + (size_t)c * 14 * 8 * 2;
    const float step = 6.283185307179586f / 14.f;
    float acc = 0.f;
    #pragma unroll 1
    for (int a = 0; a < 14; ++a) {
        #pragma unroll
        for (int b = 0; b < 8; ++b) {
            float wb = (b == 0 || b == 7) ? 1.f : 2.f;
            float wr = wc[(a * 8 + b) * 2 + 0];
            float wi = wc[(a * 8 + b) * 2 + 1];
            float ph = step * (float)(a * u + b * v);
            float s, co;
            __sincosf(ph, &s, &co);
            acc += wb * (wr * co - wi * s);
        }
    }
    kspat[idx] = acc * (1.f / 196.f);
}

// Depthwise 3x3, SAME zero padding, channels 0..383 -> feat rows [c][p] (f16)
__global__ void dwconv_kernel(const float* __restrict__ x,
                              const float* __restrict__ w,
                              _Float16* __restrict__ feat) {
    int idx = blockIdx.x * 256 + threadIdx.x;       // img*384*196 + c*196 + pos
    if (idx >= B_IMG * HALF * HW) return;
    int pos = idx % HW;
    int t   = idx / HW;
    int c   = t % HALF;
    int img = t / HALF;
    int h = pos / 14, ww = pos % 14;
    const float* xs = x + ((size_t)img * DIM + c) * HW;
    const float* wk = w + (size_t)c * 9;
    float acc = 0.f;
    #pragma unroll
    for (int dy = -1; dy <= 1; ++dy)
        #pragma unroll
        for (int dx = -1; dx <= 1; ++dx) {
            int hh = h + dy, cc = ww + dx;
            if (hh >= 0 && hh < 14 && cc >= 0 && cc < 14)
                acc += xs[hh * 14 + cc] * wk[(dy + 1) * 3 + (dx + 1)];
        }
    feat[(size_t)c * P_TOT + img * HW + pos] = (_Float16)acc;
}

// Per-channel 14x14 circular convolution, channels 384..767 -> feat (f16)
__global__ __launch_bounds__(256)
void circconv_kernel(const float* __restrict__ x,
                     const float* __restrict__ kspat,
                     _Float16* __restrict__ feat) {
    __shared__ float xs[HW];
    __shared__ float ks[HW];
    int c   = blockIdx.x;        // 0..383 (half2 channel)
    int img = blockIdx.y;        // 0..127
    int tid = threadIdx.x;
    const float* xsrc = x + ((size_t)img * DIM + HALF + c) * HW;
    if (tid < HW) { xs[tid] = xsrc[tid]; ks[tid] = kspat[(size_t)c * HW + tid]; }
    __syncthreads();
    if (tid < HW) {
        int ph = tid / 14, pw = tid % 14;
        float acc = 0.f;
        #pragma unroll 1
        for (int dh = 0; dh < 14; ++dh) {
            int qh = ph - dh; if (qh < 0) qh += 14;
            const float* krow = &ks[dh * 14];
            const float* xrow = &xs[qh * 14];
            #pragma unroll
            for (int dw = 0; dw < 14; ++dw) {
                int qw = pw - dw; if (qw < 0) qw += 14;
                acc += krow[dw] * xrow[qw];
            }
        }
        feat[((size_t)(HALF + c)) * P_TOT + img * HW + tid] = (_Float16)acc;
    }
}

// --------------------------- WMMA projection GEMM --------------------------
// out[o][p] = sum_c WprojH[o][c] * feat[c][p] + bias[o]
// Block tile: 128(O) x 128(P); 8 wave32s -> each wave 64x32 (4x2 frags of 16x16)
// A (f16 weights) staged per K-step by the Tensor Data Mover with LDS padding
// (pad_interval=16 DWORDs, pad_amount=4 DWORDs -> row stride 40 f16); B staged
// manually with a transpose into [n][k]. Fuses bias + BN partial statistics.

#define LDS_STRIDE 40   // 32 k + 8 pad (f16) ; row = 80 B, 16B aligned

__global__ __launch_bounds__(256)
void gemm_proj_kernel(const _Float16* __restrict__ WprojH,
                      const _Float16* __restrict__ feat,
                      const float* __restrict__ bias,
                      float* __restrict__ out,          // (B,DIM,H,W) raw
                      float* __restrict__ gsum,
                      float* __restrict__ gsq) {
    __shared__ __align__(16) _Float16 As[128 * LDS_STRIDE];
    __shared__ __align__(16) _Float16 Bs[128 * LDS_STRIDE];
    __shared__ float csum[128];
    __shared__ float csq[128];

    const int tid  = threadIdx.x;
    const int lane = tid & 31;
    const int wave = tid >> 5;
    const int p0   = blockIdx.x * 128;   // position tile (196 tiles, exact)
    const int o0   = blockIdx.y * 128;   // out-channel tile (6 tiles, exact)

    if (tid < 128) { csum[tid] = 0.f; csq[tid] = 0.f; }

    const int wm = wave & 1;             // 0..1 : 64-row slab
    const int wn = wave >> 1;            // 0..3 : 32-col slab

    v8f acc[4][2];
    const v8f vzero = {0.f, 0.f, 0.f, 0.f, 0.f, 0.f, 0.f, 0.f};
    #pragma unroll
    for (int i = 0; i < 4; ++i)
        #pragma unroll
        for (int j = 0; j < 2; ++j) acc[i][j] = vzero;

    union HV { v16h v; h8 h[2]; };

    // staging coordinates (fixed per thread)
    const int bk = tid >> 3;             // B: k row 0..31
    const int bn = (tid & 7) * 16;       // B: 16 columns
#if !HAVE_TDM
    const int am = tid >> 1;             // A: row 0..127
    const int ak = (tid & 1) * 16;       // A: k sub-chunk 0/16
#endif

#if HAVE_TDM
    // ---- TDM descriptor group 1 (constant across K-steps) ----
    // data_size=1 (2B), pad_enable=1, pad_interval=3 (16 DW = 32 f16),
    // pad_amount=3 (4 DW = 8 f16)  => LDS row stride 40 f16.
    // tensor_dim0 = tensor_dim1 = 768, tile_dim0 = 32 (K), tile_dim1 = 128 (M),
    // tensor_dim0_stride = 768 elements.
    union { unsigned int w[8]; i32x8 v; } g1u;
    g1u.w[0] = (1u << 16) | (1u << 20) | (3u << 22) | (3u << 25);
    g1u.w[1] = ((unsigned)K_DIM & 0xFFFFu) << 16;           // tensor_dim0[15:0]
    g1u.w[2] = (((unsigned)K_DIM >> 16) & 0xFFFFu)          // tensor_dim0[31:16]
             | (((unsigned)K_DIM & 0xFFFFu) << 16);         // tensor_dim1[15:0]
    g1u.w[3] = (((unsigned)K_DIM >> 16) & 0xFFFFu)          // tensor_dim1[31:16]
             | (32u << 16);                                 // tile_dim0 = 32
    g1u.w[4] = 128u;                                        // tile_dim1=128, tile_dim2=0
    g1u.w[5] = (unsigned)K_DIM;                             // dim0_stride[31:0]
    g1u.w[6] = 0u;                                          // dim0_stride[47:32], dim1_stride lo
    g1u.w[7] = 0u;
    const unsigned int lds_a_base = (unsigned int)(size_t)(&As[0]);
#endif

    for (int k0 = 0; k0 < K_DIM; k0 += 32) {
        // ---- stage A (f16 weight tile) ----
#if HAVE_TDM
        if (wave == 0) {
            unsigned long long ga = (unsigned long long)(size_t)WprojH
                + ((unsigned long long)o0 * K_DIM + (unsigned)k0) * 2ull;
            union { unsigned int w[4]; u32x4 v; } g0u;
            g0u.w[0] = 1u;                                   // count=1
            g0u.w[1] = lds_a_base;                           // lds_addr
            g0u.w[2] = (unsigned int)ga;                     // global_addr[31:0]
            g0u.w[3] = (unsigned int)((ga >> 32) & 0x1FFFFFFull)
                     | (2u << 30);                           // addr[56:32], type=2
            i32x4 z4 = {0, 0, 0, 0};
#if __clang_major__ >= 23
            i32x8 z8 = {0, 0, 0, 0, 0, 0, 0, 0};
            __builtin_amdgcn_tensor_load_to_lds(g0u.v, g1u.v, z4, z4, z8, 0);
#else
            __builtin_amdgcn_tensor_load_to_lds(g0u.v, g1u.v, z4, z4, 0);
#endif
        }
#else
        {
            const _Float16* src = WprojH + (size_t)(o0 + am) * K_DIM + k0 + ak;
            h8 a0 = *(const h8*)(src + 0);
            h8 a1 = *(const h8*)(src + 8);
            *(h8*)(&As[am * LDS_STRIDE + ak + 0]) = a0;
            *(h8*)(&As[am * LDS_STRIDE + ak + 8]) = a1;
        }
#endif
        // ---- stage B (feat tile, f16), transpose to LDS layout [n][k] ----
        {
            const _Float16* src = feat + (size_t)(k0 + bk) * P_TOT + p0 + bn;
            h8 b0 = *(const h8*)(src + 0);
            h8 b1 = *(const h8*)(src + 8);
            #pragma unroll
            for (int j = 0; j < 8; ++j) Bs[(bn + j) * LDS_STRIDE + bk]     = b0[j];
            #pragma unroll
            for (int j = 0; j < 8; ++j) Bs[(bn + 8 + j) * LDS_STRIDE + bk] = b1[j];
            // prefetch next K tile of feat (emits global_prefetch_b8)
            int k1 = k0 + 32; if (k1 >= K_DIM) k1 = 0;
            __builtin_prefetch(feat + (size_t)(k1 + bk) * P_TOT + p0 + bn, 0, 3);
        }
#if HAVE_TDM
        if (wave == 0) __builtin_amdgcn_s_wait_tensorcnt(0);
#endif
        __syncthreads();

        // ---- fragment loads per ISA VGPR layouts ----
        const int mrow = lane & 15;
        const int kb   = (lane >> 4) * 8;    // A: K-half base (0 / 8)
        const int kbs  = (lane >> 4) * 16;   // unused (B loads use kb, as A does)
        HV a[4], b[2];
        #pragma unroll
        for (int i = 0; i < 4; ++i) {
            int m = wm * 64 + i * 16 + mrow;
            a[i].h[0] = *(const h8*)(&As[m * LDS_STRIDE + kb]);       // K kb..kb+7
            a[i].h[1] = *(const h8*)(&As[m * LDS_STRIDE + kb + 16]);  // K kb+16..+23
        }
        #pragma unroll
        for (int j = 0; j < 2; ++j) {
            int n = wn * 32 + j * 16 + mrow;
            b[j].h[0] = *(const h8*)(&Bs[n * LDS_STRIDE + kb]);       // K kb..kb+7
            b[j].h[1] = *(const h8*)(&Bs[n * LDS_STRIDE + kb + 16]);  // K kb+16..+23
        }
        #pragma unroll
        for (int i = 0; i < 4; ++i)
            #pragma unroll
            for (int j = 0; j < 2; ++j)
                acc[i][j] = __builtin_amdgcn_wmma_f32_16x16x32_f16(
                    false, a[i].v, false, b[j].v, (short)0, acc[i][j], false, false);
        __syncthreads();
    }

    // ---- epilogue: bias, store to (B,C,H,W), fused BN partial stats ----
    const int nn    = lane & 15;
    const int mbase = (lane >> 4) * 8;      // D: lanes 16-31 hold M=8..15
    #pragma unroll
    for (int i = 0; i < 4; ++i) {
        int mt = wm * 64 + i * 16;
        #pragma unroll
        for (int j = 0; j < 2; ++j) {
            int p   = p0 + wn * 32 + j * 16 + nn;
            int img = p / HW;
            int pos = p % HW;
            #pragma unroll
            for (int r = 0; r < 8; ++r) {
                int m = mbase + r;
                int o = o0 + mt + m;
                float v = acc[i][j][r] + bias[o];
                out[((size_t)img * DIM + o) * HW + pos] = v;
                atomicAdd(&csum[mt + m], v);
                atomicAdd(&csq[mt + m], v * v);
            }
        }
    }
    __syncthreads();
    if (tid < 128) {
        atomicAdd(&gsum[o0 + tid], csum[tid]);
        atomicAdd(&gsq[o0 + tid],  csq[tid]);
    }
}

// ------------------------------- BatchNorm ---------------------------------

__global__ void bn_stats_kernel(const float* __restrict__ gsum,
                                const float* __restrict__ gsq,
                                const float* __restrict__ gamma,
                                const float* __restrict__ beta,
                                float* __restrict__ scale,
                                float* __restrict__ shiftv) {
    int o = blockIdx.x * 256 + threadIdx.x;
    if (o >= DIM) return;
    const float n = (float)P_TOT;
    float mean = gsum[o] / n;
    float var  = gsq[o] / n - mean * mean;
    float inv  = rsqrtf(var + 1e-5f);
    float sc   = gamma[o] * inv;
    scale[o]  = sc;
    shiftv[o] = beta[o] - mean * sc;
}

__global__ void bn_apply_kernel(float* __restrict__ out,
                                const float* __restrict__ scale,
                                const float* __restrict__ shiftv) {
    size_t i = (size_t)blockIdx.x * 256 + threadIdx.x;   // over B*DIM*HW (exact)
    int o = (int)((i / HW) % DIM);
    out[i] = out[i] * scale[o] + shiftv[o];
}

// ------------------------------- launcher ----------------------------------

extern "C" void kernel_launch(void* const* d_in, const int* in_sizes, int n_in,
                              void* d_out, int out_size, void* d_ws, size_t ws_size,
                              hipStream_t stream) {
    const float* x      = (const float*)d_in[0];   // (128,768,14,14)
    const float* dw_w   = (const float*)d_in[1];   // (384,1,3,3)
    const float* cw     = (const float*)d_in[2];   // (384,14,8,2)
    const float* proj_w = (const float*)d_in[3];   // (768,768,1,1)
    const float* proj_b = (const float*)d_in[4];   // (768,)
    const float* gamma  = (const float*)d_in[5];   // (768,)
    const float* beta   = (const float*)d_in[6];   // (768,)
    float* out = (float*)d_out;                    // (128,768,14,14)

    char* ws = (char*)d_ws;
    _Float16* feat = (_Float16*)ws;                                // 38,535,168 B
    float* kspat   = (float*)(ws + (size_t)DIM * P_TOT * 2);       // 301,056 B
    float* gsum    = kspat + HALF * HW;
    float* gsq     = gsum + DIM;
    float* scale   = gsq + DIM;
    float* shiftv  = scale + DIM;
    _Float16* wprojh = (_Float16*)(shiftv + DIM);                  // 1,179,648 B

    init_stats_kernel<<<3, 256, 0, stream>>>(gsum, gsq);
    convert_wproj_kernel<<<(DIM * K_DIM) / 256, 256, 0, stream>>>(proj_w, wprojh);
    build_spatial_kernel<<<(HALF * HW + 255) / 256, 256, 0, stream>>>(cw, kspat);
    dwconv_kernel<<<(B_IMG * HALF * HW + 255) / 256, 256, 0, stream>>>(x, dw_w, feat);
    circconv_kernel<<<dim3(HALF, B_IMG), 256, 0, stream>>>(x, kspat, feat);
    gemm_proj_kernel<<<dim3(P_TOT / 128, DIM / 128), 256, 0, stream>>>(
        wprojh, feat, proj_b, out, gsum, gsq);
    bn_stats_kernel<<<3, 256, 0, stream>>>(gsum, gsq, gamma, beta, scale, shiftv);
    bn_apply_kernel<<<(B_IMG * DIM * HW) / 256, 256, 0, stream>>>(out, scale, shiftv);
}
